// wrdgen_71038759076095
// MI455X (gfx1250) — compile-verified, matrix operand order checked by reading, not run on hardware
//
#include <hip/hip_runtime.h>
#include <hip/hip_bf16.h>

// ---------------------------------------------------------------------------
// Word-generator RNN on gfx1250.
//   - 24 timesteps x 6 layers, each layer = one fused WMMA GEMM
//     [512 x 800(f16)] @ [800 x 1280(f16)] -> gate epilogue (sigmoid/tanh)
//   - weights packed once per call into f16 [N][K] (B loaded as W^T),
//     N layout: cols 0..999   = interleaved (sw_j, mem_j) pairs (j<500)
//               cols 1000..1023 = zero pad
//               cols 1024..1273 = sv (x) columns
//               cols 1274..1279 = zero pad
//   - activation buffer A[512][800] f16: cols 0..249 = x, 250..749 = m,
//     750..779 = one-hot letter, 780..799 = zero-weight padding
//   - f32 master copy of m for precision across 24 recurrent updates
// ---------------------------------------------------------------------------

typedef __attribute__((ext_vector_type(16))) _Float16 v16h;
typedef __attribute__((ext_vector_type(8)))  _Float16 v8h;
typedef __attribute__((ext_vector_type(8)))  float    v8f;

#define BATCH 512
#define TSTEPS 24
#define VOC 30
#define KPAD 800      // padded K (750/780 -> 800), 25 chunks of 32
#define NPAD 1280     // padded N (1250 -> 1280), 80 tiles of 16
#define BM 64         // block tile M
#define BN 128        // block tile N
#define KB 160        // K per LDS stage (5 wmma k-chunks)
#define NSTAGE (KPAD / KB)   // 5
#define LDSS (KB + 8)        // padded LDS row stride in halfs (336B, conflict-free)

// ---------------------------------------------------------------------------
// WMMA fragment load from LDS.
// 16-bit A-matrix 16x32 layout (ISA 7.12.2): lane L -> row = L%16,
// klo = (L<16)?0:8; element e<8 -> K = klo+e ; e>=8 -> K = klo+8+e.
// So a fragment is two contiguous 16B reads at +0 and +32B.
// B is loaded from W^T ([N][K]) with the identical pattern (lane -> N row).
// ---------------------------------------------------------------------------
__device__ __forceinline__ v16h ldfrag(const _Float16* p) {
  v8h lo = *(const v8h*)p;
  v8h hi = *(const v8h*)(p + 16);
  v16h r;
#pragma unroll
  for (int i = 0; i < 8; ++i) { r[i] = lo[i]; r[i + 8] = hi[i]; }
  return r;
}

// ---------------------------------------------------------------------------
// Fused GEMM + gate kernel.
//   z = A_in @ W^T + bias     (WMMA f16 -> f32)
//   cols <1000 (pairs): s=sigmoid(z_sw); m_new = m*s + tanh(z_mem)*(1-s)
//   cols 1024..1273:    x_new = tanh(z_sv)
// Writes m_new to f32 master + f16 A_out[250..749], x_new to A_out[0..249].
// ---------------------------------------------------------------------------
__global__ __launch_bounds__(256, 1)
void gemm_gate(const _Float16* __restrict__ Ain,   // [512][KPAD]
               const _Float16* __restrict__ Wt,    // [NPAD][KPAD]
               const float*    __restrict__ bias,  // [NPAD]
               const float*    __restrict__ m_in,  // [512][500]
               float*          __restrict__ m_out, // [512][500]
               _Float16*       __restrict__ Aout)  // [512][KPAD]
{
  __shared__ __align__(16) _Float16 sA[BM][LDSS];
  __shared__ __align__(16) _Float16 sB[BN][LDSS];

  const int tid    = threadIdx.x;
  const int m_base = blockIdx.x * BM;   // 0..7
  const int n_base = blockIdx.y * BN;   // 0..9

  const int wave = tid >> 5;
  const int lane = tid & 31;
  const int wa   = wave & 1;    // M tile pair: {2wa, 2wa+1}
  const int wb   = wave >> 1;   // N tile pair: {2wb, 2wb+1}

  const int klo  = (lane < 16) ? 0 : 8;
  const int rrow = lane & 15;

  v8f acc[2][2] = {};

  for (int s = 0; s < NSTAGE; ++s) {
    const int koff = s * KB;
    // stage A block: 64 x 160 halfs, 16B vectors
#pragma unroll
    for (int v = tid; v < BM * (KB / 8); v += 256) {
      int r = v / (KB / 8), c = v % (KB / 8);
      *(uint4*)&sA[r][c * 8] =
          *(const uint4*)&Ain[(size_t)(m_base + r) * KPAD + koff + c * 8];
    }
    // stage B block (W^T rows): 128 x 160 halfs
#pragma unroll
    for (int v = tid; v < BN * (KB / 8); v += 256) {
      int r = v / (KB / 8), c = v % (KB / 8);
      *(uint4*)&sB[r][c * 8] =
          *(const uint4*)&Wt[(size_t)(n_base + r) * KPAD + koff + c * 8];
    }
    __syncthreads();

#pragma unroll
    for (int kc = 0; kc < KB / 32; ++kc) {
      const int c0 = kc * 32 + klo;
      v16h af[2], bf[2];
#pragma unroll
      for (int i = 0; i < 2; ++i) {
        af[i] = ldfrag(&sA[(2 * wa + i) * 16 + rrow][c0]);
        bf[i] = ldfrag(&sB[(2 * wb + i) * 16 + rrow][c0]);
      }
#pragma unroll
      for (int i = 0; i < 2; ++i)
#pragma unroll
        for (int j = 0; j < 2; ++j)
          acc[i][j] = __builtin_amdgcn_wmma_f32_16x16x32_f16(
              false, af[i], false, bf[j], (short)0, acc[i][j], false, false);
    }
    __syncthreads();
  }

  // ---- gate epilogue ----
#pragma unroll
  for (int i = 0; i < 2; ++i) {
    const int m0 = m_base + (2 * wa + i) * 16 + ((lane >> 4) ? 8 : 0);
#pragma unroll
    for (int j = 0; j < 2; ++j) {
      const int n  = n_base + (2 * wb + j) * 16 + (lane & 15);
      const float bn = bias[n];
      v8f d = acc[i][j];
#pragma unroll
      for (int e = 0; e < 8; ++e) {
        const int row = m0 + e;
        float z  = d[e] + bn;
        float zp = __shfl_xor(z, 1, 32);   // partner column (sw <-> mem)
        if (n < 1000) {
          if ((n & 1) == 0) {              // even lane owns the pair
            float sgate = 1.0f / (1.0f + __expf(-z));
            float tmem  = tanhf(zp);
            int   jj    = n >> 1;
            float mo    = m_in[(size_t)row * 500 + jj];
            float mn    = mo * sgate + tmem * (1.0f - sgate);
            m_out[(size_t)row * 500 + jj]        = mn;
            Aout[(size_t)row * KPAD + 250 + jj]  = (_Float16)mn;
          }
        } else if (n >= 1024 && n < 1274) {
          Aout[(size_t)row * KPAD + (n - 1024)] = (_Float16)tanhf(z);
        }
      }
    }
  }
}

// ---------------------------------------------------------------------------
// Weight packing: f32 [K][Ncols] -> f16 W^T [NPAD][KPAD] with interleaved N.
// ---------------------------------------------------------------------------
__global__ void pack_w1(const float* __restrict__ w_sw, const float* __restrict__ w_mem,
                        const float* __restrict__ w_sv, const float* __restrict__ b_sw,
                        const float* __restrict__ b_mem, const float* __restrict__ b_sv,
                        _Float16* __restrict__ Wt, float* __restrict__ bias)
{
  long i = (long)blockIdx.x * blockDim.x + threadIdx.x;
  if (i >= (long)NPAD * KPAD) return;
  int n = (int)(i / KPAD), k = (int)(i % KPAD);
  float v = 0.0f;
  if (k < 780) {
    if (n < 1000) {
      int j = n >> 1;
      v = (n & 1) ? w_mem[(size_t)k * 500 + j] : w_sw[(size_t)k * 500 + j];
    } else if (n >= 1024 && n < 1274) {
      v = w_sv[(size_t)k * 250 + (n - 1024)];
    }
  }
  Wt[i] = (_Float16)v;
  if (k == 0) {
    float bv = 0.0f;
    if (n < 1000)                  bv = (n & 1) ? b_mem[n >> 1] : b_sw[n >> 1];
    else if (n >= 1024 && n < 1274) bv = b_sv[n - 1024];
    bias[n] = bv;
  }
}

__global__ void pack_wL(const float* __restrict__ w_sw, const float* __restrict__ w_mem,
                        const float* __restrict__ w_sv, const float* __restrict__ b_sw,
                        const float* __restrict__ b_mem, const float* __restrict__ b_sv,
                        _Float16* __restrict__ Wt, float* __restrict__ bias)
{
  const long per = (long)NPAD * KPAD;
  long i = (long)blockIdx.x * blockDim.x + threadIdx.x;
  if (i >= 5 * per) return;
  int  l = (int)(i / per);
  long r = i % per;
  int n = (int)(r / KPAD), k = (int)(r % KPAD);
  float v = 0.0f;
  if (k < 750) {
    if (n < 1000) {
      int j = n >> 1;
      v = (n & 1) ? w_mem[((size_t)l * 750 + k) * 500 + j]
                  : w_sw[((size_t)l * 750 + k) * 500 + j];
    } else if (n >= 1024 && n < 1274) {
      v = w_sv[((size_t)l * 750 + k) * 250 + (n - 1024)];
    }
  }
  Wt[i] = (_Float16)v;
  if (k == 0) {
    float bv = 0.0f;
    if (n < 1000)                   bv = (n & 1) ? b_mem[l * 500 + (n >> 1)] : b_sw[l * 500 + (n >> 1)];
    else if (n >= 1024 && n < 1274) bv = b_sv[l * 250 + (n - 1024)];
    bias[l * NPAD + n] = bv;
  }
}

// zero x,m columns of A0 and the f32 m state
__global__ void init_state(_Float16* __restrict__ A0, float* __restrict__ m0)
{
  int i = blockIdx.x * blockDim.x + threadIdx.x;
  if (i < BATCH * 750) {
    int r = i / 750, c = i % 750;
    A0[(size_t)r * KPAD + c] = (_Float16)0;
  }
  if (i < BATCH * 500) m0[i] = 0.0f;
}

// write one-hot(letter[t]) into A0 cols 750..779
__global__ void onehot_step(const int* __restrict__ letters, _Float16* __restrict__ A0, int t)
{
  int b = blockIdx.x * blockDim.x + threadIdx.x;
  if (b >= BATCH) return;
  int v = letters[b * TSTEPS + t];
  _Float16* p = A0 + (size_t)b * KPAD + 750;
#pragma unroll
  for (int i = 0; i < VOC; ++i) p[i] = (_Float16)0;
  if (v >= 0 && v < VOC) p[v] = (_Float16)1;
}

// ---------------------------------------------------------------------------
// Prediction head (tiny: ~0.3% of total FLOPs) in plain f32.
// ---------------------------------------------------------------------------
__global__ void head_first(const _Float16* __restrict__ A, const float* __restrict__ W,
                           const float* __restrict__ b, float* __restrict__ out)
{
  int idx = blockIdx.x * blockDim.x + threadIdx.x;
  if (idx >= BATCH * 450) return;
  int r = idx / 450, n = idx % 450;
  float acc = b[n];
  for (int k = 0; k < 750; ++k)
    acc += (float)A[(size_t)r * KPAD + k] * W[(size_t)k * 450 + n];
  out[idx] = tanhf(acc);
}

__global__ void head_mid(const float* __restrict__ in, const float* __restrict__ W,
                         const float* __restrict__ b, float* __restrict__ out, int K, int N)
{
  int idx = blockIdx.x * blockDim.x + threadIdx.x;
  if (idx >= BATCH * N) return;
  int r = idx / N, n = idx % N;
  float acc = b[n];
  for (int k = 0; k < K; ++k)
    acc += in[(size_t)r * K + k] * W[(size_t)k * N + n];
  out[idx] = tanhf(acc);
}

__global__ void head_final(const float* __restrict__ in, const float* __restrict__ W,
                           const float* __restrict__ b, float* __restrict__ out)
{
  __shared__ float lg[VOC];
  __shared__ float ssum;
  int r = blockIdx.x;
  int lane = threadIdx.x;
  if (lane < VOC) {
    float acc = b[lane];
    for (int k = 0; k < 100; ++k)
      acc += in[(size_t)r * 100 + k] * W[(size_t)k * VOC + lane];
    lg[lane] = acc;
  }
  __syncthreads();
  if (lane == 0) {
    float mx = -1e30f;
    for (int i = 0; i < VOC; ++i) mx = fmaxf(mx, lg[i]);
    float s = 0.0f;
    for (int i = 0; i < VOC; ++i) { lg[i] = __expf(lg[i] - mx); s += lg[i]; }
    ssum = s;
  }
  __syncthreads();
  if (lane < VOC) out[(size_t)r * VOC + lane] = lg[lane] / ssum;
}

// ---------------------------------------------------------------------------
extern "C" void kernel_launch(void* const* d_in, const int* in_sizes, int n_in,
                              void* d_out, int out_size, void* d_ws, size_t ws_size,
                              hipStream_t stream)
{
  const int*   letters = (const int*)d_in[0];
  const float* w1_sv = (const float*)d_in[1];
  const float* b1_sv = (const float*)d_in[2];
  const float* w1_mem = (const float*)d_in[3];
  const float* b1_mem = (const float*)d_in[4];
  const float* w1_sw = (const float*)d_in[5];
  const float* b1_sw = (const float*)d_in[6];
  const float* w_sv  = (const float*)d_in[7];
  const float* b_sv  = (const float*)d_in[8];
  const float* w_mem = (const float*)d_in[9];
  const float* b_mem = (const float*)d_in[10];
  const float* w_sw  = (const float*)d_in[11];
  const float* b_sw  = (const float*)d_in[12];
  const float* wp1 = (const float*)d_in[13]; const float* bp1 = (const float*)d_in[14];
  const float* wp2 = (const float*)d_in[15]; const float* bp2 = (const float*)d_in[16];
  const float* wp3 = (const float*)d_in[17]; const float* bp3 = (const float*)d_in[18];
  const float* wp4 = (const float*)d_in[19]; const float* bp4 = (const float*)d_in[20];
  const float* wp5 = (const float*)d_in[21]; const float* bp5 = (const float*)d_in[22];

  // ---- workspace carve-up (256B aligned) ----
  char* ws = (char*)d_ws;
  size_t off = 0;
  auto carve = [&](size_t bytes) -> char* {
    char* p = ws + off;
    off = (off + bytes + 255) & ~(size_t)255;
    return p;
  };
  const size_t wbytes = (size_t)NPAD * KPAD * sizeof(_Float16);
  _Float16* Wt1   = (_Float16*)carve(wbytes);
  _Float16* WtL   = (_Float16*)carve(5 * wbytes);
  float*    bias1 = (float*)carve(NPAD * sizeof(float));
  float*    biasL = (float*)carve(5 * NPAD * sizeof(float));
  _Float16* Abuf0 = (_Float16*)carve((size_t)BATCH * KPAD * sizeof(_Float16));
  _Float16* Abuf1 = (_Float16*)carve((size_t)BATCH * KPAD * sizeof(_Float16));
  float*    mbuf0 = (float*)carve((size_t)BATCH * 500 * sizeof(float));
  float*    mbuf1 = (float*)carve((size_t)BATCH * 500 * sizeof(float));
  float*    y1    = (float*)carve((size_t)BATCH * 450 * sizeof(float));
  float*    y2    = (float*)carve((size_t)BATCH * 300 * sizeof(float));
  float*    y3    = (float*)carve((size_t)BATCH * 200 * sizeof(float));
  float*    y4    = (float*)carve((size_t)BATCH * 100 * sizeof(float));
  (void)ws_size; (void)n_in; (void)in_sizes; (void)out_size;

  _Float16* Abuf[2] = {Abuf0, Abuf1};
  float*    mbuf[2] = {mbuf0, mbuf1};

  // ---- pack weights (once per call; cheap vs 144 GEMMs) ----
  {
    long n1 = (long)NPAD * KPAD;
    pack_w1<<<(unsigned)((n1 + 255) / 256), 256, 0, stream>>>(
        w1_sw, w1_mem, w1_sv, b1_sw, b1_mem, b1_sv, Wt1, bias1);
    long n5 = 5L * NPAD * KPAD;
    pack_wL<<<(unsigned)((n5 + 255) / 256), 256, 0, stream>>>(
        w_sw, w_mem, w_sv, b_sw, b_mem, b_sv, WtL, biasL);
  }

  // ---- init state ----
  init_state<<<(BATCH * 750 + 255) / 256, 256, 0, stream>>>(Abuf0, mbuf0);

  // ---- recurrent scan: 24 steps x 6 fused GEMM+gate layers ----
  dim3 ggrid(BATCH / BM, NPAD / BN);   // 8 x 10
  int pin = 0;
  for (int t = 0; t < TSTEPS; ++t) {
    onehot_step<<<(BATCH + 255) / 256, 256, 0, stream>>>(letters, Abuf0, t);
    for (int l = 0; l < 6; ++l) {
      const _Float16* Wt = (l == 0) ? Wt1 : WtL + (size_t)(l - 1) * NPAD * KPAD;
      const float*    bs = (l == 0) ? bias1 : biasL + (size_t)(l - 1) * NPAD;
      gemm_gate<<<ggrid, 256, 0, stream>>>(Abuf[pin], Wt, bs,
                                           mbuf[pin], mbuf[1 - pin], Abuf[1 - pin]);
      pin ^= 1;
    }
  }
  // 144 layers (even) -> final state lives in Abuf[0]

  // ---- head ----
  head_first<<<(BATCH * 450 + 255) / 256, 256, 0, stream>>>(Abuf0, wp1, bp1, y1);
  head_mid<<<(BATCH * 300 + 255) / 256, 256, 0, stream>>>(y1, wp2, bp2, y2, 450, 300);
  head_mid<<<(BATCH * 200 + 255) / 256, 256, 0, stream>>>(y2, wp3, bp3, y3, 300, 200);
  head_mid<<<(BATCH * 100 + 255) / 256, 256, 0, stream>>>(y3, wp4, bp4, y4, 200, 100);
  head_final<<<BATCH, 32, 0, stream>>>(y4, wp5, bp5, (float*)d_out);
}
